// CheemsNonWoAttn_36636071034961
// MI455X (gfx1250) — matrix operand order repaired and checked by reading, not compiled, not run on hardware
//
#include <hip/hip_runtime.h>
#include <hip/hip_bf16.h>

typedef __attribute__((ext_vector_type(16))) _Float16 v16h;
typedef __attribute__((ext_vector_type(8)))  _Float16 h8;
typedef __attribute__((ext_vector_type(8)))  float    v8f;
typedef __attribute__((ext_vector_type(4)))  float    f4;

#define SLEN 2048
#define HIDN 2048
#define NHD  32
#define HKVN 8
#define HDIM 64

#define WMMA_F16(A, B, C) \
  __builtin_amdgcn_wmma_f32_16x16x32_f16(false, (A), false, (B), (short)0, (C), false, false)

__device__ __forceinline__ h8 ldh8(const _Float16* p) {
  return *reinterpret_cast<const h8*>(p);
}
__device__ __forceinline__ v16h cat16(h8 lo, h8 hi) {
  return __builtin_shufflevector(lo, hi, 0, 1, 2, 3, 4, 5, 6, 7, 8, 9, 10, 11, 12, 13, 14, 15);
}
// A-fragment from a row-major f16 row pointer (row = lane&15): two b128 loads
__device__ __forceinline__ v16h ldA(const _Float16* row, int hf) {
  return cat16(ldh8(row + hf * 8), ldh8(row + 16 + hf * 8));
}
// B-fragment from col-major column pointer: K = hf*8..+7 and 16+hf*8..+7 (matches ldA)
__device__ __forceinline__ v16h ldB(const _Float16* col, int hf) {
  return cat16(ldh8(col + hf * 8), ldh8(col + 16 + hf * 8));
}

// ---------------------------------------------------------------------------
// Kernel 1: X (f32 [4096,2048]) -> Xh (f16, same layout).  8 elems/thread.
// ---------------------------------------------------------------------------
__global__ void __launch_bounds__(256)
conv_x_kernel(const float* __restrict__ X, _Float16* __restrict__ Xh) {
  const size_t i = ((size_t)blockIdx.x * 256 + threadIdx.x) * 8;
  f4 a = *reinterpret_cast<const f4*>(X + i);
  f4 b = *reinterpret_cast<const f4*>(X + i + 4);
  h8 o;
#pragma unroll
  for (int j = 0; j < 4; ++j) { o[j] = (_Float16)a[j]; o[4 + j] = (_Float16)b[j]; }
  *reinterpret_cast<h8*>(Xh + i) = o;
}

// ---------------------------------------------------------------------------
// Kernel 2: Wt[n][k] (f16 [3072,2048]) = concat(Wq,Wk,Wv)[k][n].
// One thread per (n, 8-wide k block).
// ---------------------------------------------------------------------------
__global__ void __launch_bounds__(256)
conv_w_kernel(const float* __restrict__ Wq, const float* __restrict__ Wk,
              const float* __restrict__ Wv, _Float16* __restrict__ Wt) {
  const int t  = blockIdx.x * 256 + threadIdx.x;   // 3072 * 256
  const int k0 = (t & 255) * 8;
  const int n  = t >> 8;
  const float* Wp; int ldn, nloc;
  if (n < 512)       { Wp = Wq; ldn = 512;  nloc = n; }
  else if (n < 1024) { Wp = Wk; ldn = 512;  nloc = n - 512; }
  else               { Wp = Wv; ldn = 2048; nloc = n - 1024; }
  h8 o;
#pragma unroll
  for (int j = 0; j < 8; ++j) o[j] = (_Float16)Wp[(size_t)(k0 + j) * ldn + nloc];
  *reinterpret_cast<h8*>(Wt + (size_t)n * HIDN + k0) = o;
}

// ---------------------------------------------------------------------------
// Kernel 3: QKV projection GEMM.  One 16x64 output tile per wave,
// K=2048 in 64 steps x 4 WMMAs (A reused).  Q/K -> f32 scratch,
// V -> f16 transposed Vt[b][head][d][seq].
// ---------------------------------------------------------------------------
__global__ void __launch_bounds__(256)
proj_gemm_kernel(const _Float16* __restrict__ Xh, const _Float16* __restrict__ Wt,
                 float* __restrict__ qf, float* __restrict__ kf,
                 _Float16* __restrict__ Vt) {
  const int lane = threadIdx.x & 31;
  const int wave = threadIdx.x >> 5;
  const int hf   = lane >> 4;
  const int lm   = lane & 15;
  const int m0   = blockIdx.x * 16;               // 0..4080
  const int n0   = (blockIdx.y * 8 + wave) * 64;  // 0..3008  (region-aligned)

  v8f acc[4] = {{}, {}, {}, {}};
  const _Float16* arow = Xh + (size_t)(m0 + lm) * HIDN;
  for (int k0 = 0; k0 < HIDN; k0 += 32) {
    const v16h a = ldA(arow + k0, hf);
#pragma unroll
    for (int c = 0; c < 4; ++c) {
      const v16h b = ldB(Wt + (size_t)(n0 + c * 16 + lm) * HIDN + k0, hf);
      acc[c] = WMMA_F16(a, b, acc[c]);
    }
  }

  if (n0 < 1024) {  // Q or K region -> f32 scratch [4096,512]
    float* Ob = (n0 < 512) ? qf : kf;
    const int nb = (n0 < 512) ? n0 : n0 - 512;
#pragma unroll
    for (int c = 0; c < 4; ++c)
#pragma unroll
      for (int r = 0; r < 8; ++r)
        Ob[(size_t)(m0 + r + 8 * hf) * 512 + nb + c * 16 + lm] = acc[c][r];
  } else {          // V region -> f16 transposed [b][head][d][seq]
    const int nv = n0 - 1024;
#pragma unroll
    for (int c = 0; c < 4; ++c) {
      const int nglob = nv + c * 16 + lm;   // head*64 + d
      const int nh = nglob >> 6, d = nglob & 63;
#pragma unroll
      for (int r = 0; r < 8; ++r) {
        const int m = m0 + r + 8 * hf;      // b*SLEN + s
        const int b = m >> 11, s = m & (SLEN - 1);
        Vt[(((size_t)b * NHD + nh) * HDIM + d) * SLEN + s] = (_Float16)acc[c][r];
      }
    }
  }
}

// ---------------------------------------------------------------------------
// Kernel 4: RoPE f32 -> f16.  One thread per (b,s,h,i<32) rotation pair.
// ---------------------------------------------------------------------------
__global__ void __launch_bounds__(256)
rope_kernel(const float* __restrict__ qf, const float* __restrict__ kf,
            _Float16* __restrict__ Qh, _Float16* __restrict__ Kh,
            const int* __restrict__ pos) {
  const int idx = blockIdx.x * 256 + threadIdx.x;  // B*S*HKV*32
  const int i = idx & 31;
  const int h = (idx >> 5) & 7;
  const int s = (idx >> 8) & (SLEN - 1);
  const int b = idx >> 19;

  const float p   = (float)pos[b * SLEN + s];
  const float ang = p * __expf(-(float)i * 0.2878231366242557f);  // ln(1e4)/32
  const float c = __cosf(ang), sn = __sinf(ang);

  const size_t base = ((size_t)(b * SLEN + s) * HKVN + h) * HDIM;
  float q1 = qf[base + i], q2 = qf[base + 32 + i];
  Qh[base + i]      = (_Float16)(q1 * c - q2 * sn);
  Qh[base + 32 + i] = (_Float16)(q2 * c + q1 * sn);
  float k1 = kf[base + i], k2 = kf[base + 32 + i];
  Kh[base + i]      = (_Float16)(k1 * c - k2 * sn);
  Kh[base + 32 + i] = (_Float16)(k2 * c + k1 * sn);
}

// ---------------------------------------------------------------------------
// Kernel 5: causal flash attention.  One wave per (b, kv-head h, group g,
// 16-query tile); scores depend only on h (reference repeats Q AND K);
// output head nh = g*8+h selects V columns.  All fragment loads are b128.
// ---------------------------------------------------------------------------
__global__ void __launch_bounds__(32)
attn_kernel(const _Float16* __restrict__ Qh, const _Float16* __restrict__ Kh,
            const _Float16* __restrict__ Vt, float* __restrict__ Out) {
  __shared__ __align__(16) _Float16 pl[16 * 32];

  const int lane = threadIdx.x;
  const int hf = lane >> 4, lm = lane & 15;
  int bid = blockIdx.x;
  const int qt = bid & 127; bid >>= 7;
  const int g  = bid & 3;   bid >>= 2;
  const int h  = bid & 7;   bid >>= 3;
  const int b  = bid;
  const int nh = g * 8 + h;

  // Q A-fragments: 16 queries x 64 dim
  const _Float16* qrow = Qh + ((size_t)(b * SLEN + qt * 16 + lm) * HKVN + h) * HDIM;
  const v16h qa0 = ldA(qrow, hf);
  const v16h qa1 = ldA(qrow + 32, hf);

  const size_t kbase = ((size_t)b * SLEN * HKVN + h) * HDIM;   // + key*512
  const size_t vbase = ((size_t)b * NHD + nh) * HDIM * (size_t)SLEN;  // + d*SLEN + key

  v8f o[4] = {{}, {}, {}, {}};
  float mrow[8], lrow[8];
#pragma unroll
  for (int r = 0; r < 8; ++r) { mrow[r] = -3.0e38f; lrow[r] = 0.0f; }

  const int qmax = qt * 16 + 15;
  for (int k0 = 0; k0 <= qmax; k0 += 32) {
    // ---- scores: 4 WMMAs, B-fragments contiguous in head-dim ----
    v8f s0 = {}, s1 = {};
    {
      const _Float16* kr0 = Kh + kbase + (size_t)(k0 + lm) * (HKVN * HDIM);
      const _Float16* kr1 = Kh + kbase + (size_t)(k0 + 16 + lm) * (HKVN * HDIM);
      s0 = WMMA_F16(qa0, ldB(kr0, hf), s0);
      s0 = WMMA_F16(qa1, ldB(kr0 + 32, hf), s0);
      s1 = WMMA_F16(qa0, ldB(kr1, hf), s1);
      s1 = WMMA_F16(qa1, ldB(kr1 + 32, hf), s1);
    }

    // ---- online softmax ----
#pragma unroll
    for (int r = 0; r < 8; ++r) {
      const int q = qt * 16 + r + 8 * hf;
      float v0 = s0[r] * 0.125f;
      float v1 = s1[r] * 0.125f;
      if (k0 + lm > q)      v0 = -3.0e38f;
      if (k0 + 16 + lm > q) v1 = -3.0e38f;
      float mx = fmaxf(v0, v1);
#pragma unroll
      for (int off = 8; off >= 1; off >>= 1)
        mx = fmaxf(mx, __shfl_xor(mx, off, 32));
      const float mnew = fmaxf(mrow[r], mx);
      const float al   = __expf(mrow[r] - mnew);
      mrow[r] = mnew;
      v0 = __expf(v0 - mnew);
      v1 = __expf(v1 - mnew);
      float rs = v0 + v1;
#pragma unroll
      for (int off = 8; off >= 1; off >>= 1)
        rs += __shfl_xor(rs, off, 32);
      lrow[r] = lrow[r] * al + rs;
#pragma unroll
      for (int c = 0; c < 4; ++c) o[c][r] *= al;
      pl[(r + 8 * hf) * 32 + lm]      = (_Float16)v0;
      pl[(r + 8 * hf) * 32 + 16 + lm] = (_Float16)v1;
    }
    __syncthreads();

    const v16h pa = ldA(&pl[lm * 32], hf);   // two ds_load_b128

    // ---- O += P(16x32) @ V(32x64), B contiguous along seq in Vt ----
#pragma unroll
    for (int c = 0; c < 4; ++c) {
      const v16h vb = ldB(Vt + vbase + (size_t)(c * 16 + lm) * SLEN + k0, hf);
      o[c] = WMMA_F16(pa, vb, o[c]);
    }
    __syncthreads();
  }

  // ---- normalize + store: out[b, q, nh*64 + d] (f32) ----
#pragma unroll
  for (int c = 0; c < 4; ++c)
#pragma unroll
    for (int r = 0; r < 8; ++r) {
      const int q = qt * 16 + r + 8 * hf;
      Out[((size_t)(b * SLEN + q) * NHD + nh) * HDIM + c * 16 + lm] = o[c][r] / lrow[r];
    }
}

// ---------------------------------------------------------------------------
extern "C" void kernel_launch(void* const* d_in, const int* in_sizes, int n_in,
                              void* d_out, int out_size, void* d_ws, size_t ws_size,
                              hipStream_t stream) {
  const float* X   = (const float*)d_in[0];
  const int*   pos = (const int*)  d_in[1];
  const float* Wq  = (const float*)d_in[2];
  const float* Wk  = (const float*)d_in[3];
  const float* Wv  = (const float*)d_in[4];
  float* out = (float*)d_out;

  // Workspace carve-up (bytes):
  char* w = (char*)d_ws;
  _Float16* Xh = (_Float16*)w;            w += (size_t)4096 * 2048 * 2;  // 16 MB
  _Float16* Wt = (_Float16*)w;            w += (size_t)3072 * 2048 * 2;  // 12 MB
  _Float16* Vt = (_Float16*)w;            w += (size_t)2 * 32 * 64 * 2048 * 2; // 16 MB
  float*    qf = (float*)w;               w += (size_t)4096 * 512 * 4;   // 8 MB
  float*    kf = (float*)w;               w += (size_t)4096 * 512 * 4;   // 8 MB
  _Float16* Qh = (_Float16*)w;            w += (size_t)4096 * 512 * 2;   // 4 MB
  _Float16* Kh = (_Float16*)w;                                           // 4 MB

  conv_x_kernel<<<4096, 256, 0, stream>>>(X, Xh);                 // 4096*2048 / 8 / 256
  conv_w_kernel<<<3072, 256, 0, stream>>>(Wq, Wk, Wv, Wt);        // 3072*256 threads
  proj_gemm_kernel<<<dim3(256, 6), 256, 0, stream>>>(Xh, Wt, qf, kf, Vt);
  rope_kernel<<<4096, 256, 0, stream>>>(qf, kf, Qh, Kh, pos);
  attn_kernel<<<2 * 8 * 4 * 128, 32, 0, stream>>>(Qh, Kh, Vt, out);
}
